// TwoDRNN_5308579578437
// MI455X (gfx1250) — matrix operand order repaired and checked by reading, not compiled, not run on hardware
//
#include <hip/hip_runtime.h>

typedef __attribute__((ext_vector_type(16))) __bf16 v16bf;
typedef __attribute__((ext_vector_type(8)))  __bf16 v8bf;
typedef __attribute__((ext_vector_type(8)))  float  v8f;
typedef __attribute__((ext_vector_type(4)))  unsigned int u32x4;
typedef __attribute__((ext_vector_type(8)))  int i32x8;
typedef __attribute__((ext_vector_type(4)))  int i32x4;
typedef __bf16 bf16;

#define BATCH 262144
#define DH    128
#define WAVES 16                    // 512 threads/block, 256 batch rows/block

// bf16 weight workspace (element offsets). Each (n_tile,k_tile) tile = 512 bf16 = 1KB,
// stored as two 512B halves: half h at tile_off + h*256 + lane*8 .. +7 holds the
// B-operand elements j = h*8..h*8+7 for that lane (16B lane stride -> good LDS banking).
// Logical B-operand element (lane, j) = W[kt*32 + (lane>=16?8:0) + (j<8?j:8+j)][nt*16 + (lane&15)]
#define U1_OFF    0
#define U2_OFF    16384
#define WHRZ_OFF  32768   // Whrz is [128,256] -> 16 n_tiles
#define WHN_OFF   65536
#define WS_ELEMS  81920   // * 2 bytes = 160 KB

__global__ __launch_bounds__(256) void prep_weights(
    const float* __restrict__ U1, const float* __restrict__ U2,
    const float* __restrict__ Whrz, const float* __restrict__ Whn,
    bf16* __restrict__ wsb)
{
    int e = blockIdx.x * 256 + threadIdx.x;
    if (e >= WS_ELEMS) return;
    const float* W; int ncols; int base;
    if (e < U2_OFF)        { W = U1;   ncols = 128; base = U1_OFF;   }
    else if (e < WHRZ_OFF) { W = U2;   ncols = 128; base = U2_OFF;   }
    else if (e < WHN_OFF)  { W = Whrz; ncols = 256; base = WHRZ_OFF; }
    else                   { W = Whn;  ncols = 128; base = WHN_OFF;  }
    int r    = e - base;
    int tile = r >> 9;             // nt*4 + kt
    int nt   = tile >> 2;
    int kt   = tile & 3;
    int idx  = r & 511;            // element within tile (split-half layout)
    int half = idx >> 8;           // 0: j=0..7, 1: j=8..15
    int lane = (idx >> 3) & 31;
    int j    = (half << 3) | (idx & 7);
    int K    = kt * 32 + ((lane & 16) ? 8 : 0) + (j < 8 ? j : 8 + j);
    int N    = nt * 16 + (lane & 15);
    wsb[e] = (bf16)W[K * ncols + N];
}

// A-operand (16x32 bf16) from a contiguous fp32 row.
// Lane holds K = kbase..kbase+7 and kbase+16..kbase+23 (kbase = 32c + 8*(lane>=16)).
__device__ inline v16bf make_a_tile(const float* __restrict__ rowptr, int kbase) {
    float4 p0 = *(const float4*)(rowptr + kbase);
    float4 p1 = *(const float4*)(rowptr + kbase + 4);
    float4 p2 = *(const float4*)(rowptr + kbase + 16);
    float4 p3 = *(const float4*)(rowptr + kbase + 20);
    v16bf a;
    a[0]=(__bf16)p0.x; a[1]=(__bf16)p0.y; a[2]=(__bf16)p0.z; a[3]=(__bf16)p0.w;
    a[4]=(__bf16)p1.x; a[5]=(__bf16)p1.y; a[6]=(__bf16)p1.z; a[7]=(__bf16)p1.w;
    a[8]=(__bf16)p2.x; a[9]=(__bf16)p2.y; a[10]=(__bf16)p2.z; a[11]=(__bf16)p2.w;
    a[12]=(__bf16)p3.x; a[13]=(__bf16)p3.y; a[14]=(__bf16)p3.z; a[15]=(__bf16)p3.w;
    return a;
}

// B-operand tile from LDS weight buffer (two ds_load_b128 per lane).
__device__ inline v16bf load_b_lds(const bf16* __restrict__ wbuf, int tile_off, int lane) {
    v8bf lo = *(const v8bf*)(wbuf + tile_off + lane * 8);
    v8bf hi = *(const v8bf*)(wbuf + tile_off + 256 + lane * 8);
    v16bf b;
    #pragma unroll
    for (int j = 0; j < 8; ++j) { b[j] = lo[j]; b[8 + j] = hi[j]; }
    return b;
}

__device__ inline float sigmoidf_fast(float x) {
    return __fdividef(1.0f, 1.0f + __expf(-x));     // exp(inf) path yields 0 correctly
}
__device__ inline float tanhf_fast(float x) {
    float xc = fminf(fmaxf(x, -10.0f), 10.0f);      // clamp: avoid inf/inf
    float t  = __expf(2.0f * xc);
    return __fdividef(t - 1.0f, t + 1.0f);
}

__global__ __launch_bounds__(32 * WAVES) void gru_fused(
    const float* __restrict__ inputs, const float* __restrict__ h1,
    const float* __restrict__ h2, const bf16* __restrict__ wsb,
    const float* __restrict__ Wi, const float* __restrict__ bi,
    const float* __restrict__ bhn, float* __restrict__ out)
{
    __shared__ __align__(16) bf16 wbuf[WS_ELEMS];          // 160 KB bf16 weights
    __shared__ float hbuf[WAVES][16][DH + 1];              // 132 KB fp32 h per wave

    const int wave = threadIdx.x >> 5;
    const int lane = threadIdx.x & 31;
    const int lrow = lane & 15;
    const int hi   = (lane >> 4) & 1;
    const int m0   = (blockIdx.x * WAVES + wave) * 16;

    // ---- TDM: each of the 16 waves DMAs a 10KB slice of bf16 weights into LDS -----
    {
        const unsigned w = (unsigned)__builtin_amdgcn_readfirstlane(threadIdx.x >> 5);
        const unsigned long long gaddr =
            (unsigned long long)(uintptr_t)wsb + (unsigned long long)w * 10240ull;
        const unsigned laddr = (unsigned)(uintptr_t)wbuf + w * 10240u;
        // D# group 0: count=1, lds_addr, global_addr[56:0], type=2
        u32x4 g0;
        g0[0] = 1u;
        g0[1] = laddr;
        g0[2] = (unsigned)gaddr;
        g0[3] = ((unsigned)(gaddr >> 32) & 0x01FFFFFFu) | (2u << 30);
        // D# group 1: data_size=8B (code 3); 1-row tensor of 1280 elems; tile = whole row
        i32x8 g1;
        g1[0] = 0x00030000;          // data_size=3, no multicast/iterate/pad
        g1[1] = (int)(1280u << 16);  // tensor_dim0[15:0] = 1280
        g1[2] = (int)(1u << 16);     // tensor_dim0 hi = 0; tensor_dim1 = 1
        g1[3] = (int)(1280u << 16);  // tile_dim0 = 1280
        g1[4] = 1;                   // tile_dim1 = 1, tile_dim2 = 0
        g1[5] = 1280;                // tensor_dim0_stride = 1280
        g1[6] = 0;
        g1[7] = 0;
        i32x4 z4 = {0, 0, 0, 0};
#if defined(__clang_major__) && __clang_major__ >= 23
        i32x8 z8 = {0, 0, 0, 0, 0, 0, 0, 0};
        __builtin_amdgcn_tensor_load_to_lds(g0, g1, z4, z4, z8, 0);
#else
        __builtin_amdgcn_tensor_load_to_lds(g0, g1, z4, z4, 0);
#endif
    }

    // ---- Overlap TDM with the wave's own A-tile global loads ----------------------
    v16bf a1[4], a2[4];
    {
        const float* r1 = h1 + (size_t)(m0 + lrow) * DH;
        const float* r2 = h2 + (size_t)(m0 + lrow) * DH;
        #pragma unroll
        for (int c = 0; c < 4; ++c) {
            int kb = 32 * c + hi * 8;
            a1[c] = make_a_tile(r1, kb);
            a2[c] = make_a_tile(r2, kb);
        }
    }
    float4 xin[8];
    #pragma unroll
    for (int r = 0; r < 8; ++r)
        xin[r] = *(const float4*)(inputs + (size_t)(m0 + r + 8*hi) * 4);

    __builtin_amdgcn_s_wait_tensorcnt(0);
    __syncthreads();   // weights resident in LDS for all waves

    // ---------------- Stage 1: h = h1@U1 + h2@U2 -----------------------------------
    #pragma unroll
    for (int t = 0; t < 8; ++t) {
        v8f acc = {};
        #pragma unroll
        for (int c = 0; c < 4; ++c) {
            acc = __builtin_amdgcn_wmma_f32_16x16x32_bf16(
                false, a1[c], false, load_b_lds(wbuf, U1_OFF + (t*4 + c)*512, lane),
                (short)0, acc, false, false);
            acc = __builtin_amdgcn_wmma_f32_16x16x32_bf16(
                false, a2[c], false, load_b_lds(wbuf, U2_OFF + (t*4 + c)*512, lane),
                (short)0, acc, false, false);
        }
        // C layout: element r -> (M = r + 8*hi, N = t*16 + lrow)
        #pragma unroll
        for (int r = 0; r < 8; ++r)
            hbuf[wave][r + 8*hi][t*16 + lrow] = acc[r];
    }
    __syncthreads();   // h consumed cross-lane through LDS

    // ---------------- Rebuild h as bf16 A-operands ----------------------------------
    v16bf ah[4];
    #pragma unroll
    for (int c = 0; c < 4; ++c) {
        int kb = 32 * c + hi * 8;
        v16bf a;
        #pragma unroll
        for (int j = 0; j < 8; ++j) a[j]     = (__bf16)hbuf[wave][lrow][kb + j];
        #pragma unroll
        for (int j = 0; j < 8; ++j) a[8 + j] = (__bf16)hbuf[wave][lrow][kb + 16 + j];
        ah[c] = a;
    }

    // ---------------- Stage 2: gates + output ---------------------------------------
    #pragma unroll
    for (int t = 0; t < 8; ++t) {
        v8f rac = {}, zac = {}, nac = {};
        #pragma unroll
        for (int c = 0; c < 4; ++c) {
            // three WMMAs back-to-back sharing the same A operand ah[c]
            rac = __builtin_amdgcn_wmma_f32_16x16x32_bf16(
                false, ah[c], false, load_b_lds(wbuf, WHRZ_OFF + ((t    )*4 + c)*512, lane),
                (short)0, rac, false, false);
            zac = __builtin_amdgcn_wmma_f32_16x16x32_bf16(
                false, ah[c], false, load_b_lds(wbuf, WHRZ_OFF + ((t + 8)*4 + c)*512, lane),
                (short)0, zac, false, false);
            nac = __builtin_amdgcn_wmma_f32_16x16x32_bf16(
                false, ah[c], false, load_b_lds(wbuf, WHN_OFF + (t*4 + c)*512, lane),
                (short)0, nac, false, false);
        }
        const int n = t * 16 + lrow;
        float wir[4], wiz[4], win[4];
        #pragma unroll
        for (int k = 0; k < 4; ++k) {
            wir[k] = Wi[k * 384 + n];
            wiz[k] = Wi[k * 384 + 128 + n];
            win[k] = Wi[k * 384 + 256 + n];
        }
        const float bir = bi[n], biz = bi[128 + n], bin_ = bi[256 + n], bh = bhn[n];

        #pragma unroll
        for (int r = 0; r < 8; ++r) {
            const float x0 = xin[r].x, x1 = xin[r].y, x2 = xin[r].z, x3 = xin[r].w;
            float ir  = bir  + x0*wir[0] + x1*wir[1] + x2*wir[2] + x3*wir[3];
            float iz  = biz  + x0*wiz[0] + x1*wiz[1] + x2*wiz[2] + x3*wiz[3];
            float inn = bin_ + x0*win[0] + x1*win[1] + x2*win[2] + x3*win[3];
            float rg = sigmoidf_fast(ir + rac[r]);
            float zg = sigmoidf_fast(iz + zac[r]);
            float hv = hbuf[wave][r + 8*hi][n];
            float ng = tanhf_fast(inn + rg * (nac[r] + bh));
            float nh = (1.0f - zg) * ng + zg * hv;
            size_t o = (size_t)(m0 + r + 8*hi) * DH + n;
            out[o] = nh;
            out[(size_t)BATCH * DH + o] = nh;   // reference returns (new_h, new_h)
        }
    }
}

extern "C" void kernel_launch(void* const* d_in, const int* in_sizes, int n_in,
                              void* d_out, int out_size, void* d_ws, size_t ws_size,
                              hipStream_t stream) {
    (void)in_sizes; (void)n_in; (void)out_size; (void)ws_size;
    const float* inputs = (const float*)d_in[0];
    const float* h1     = (const float*)d_in[1];
    const float* h2     = (const float*)d_in[2];
    const float* U1     = (const float*)d_in[3];
    const float* U2     = (const float*)d_in[4];
    const float* Wi     = (const float*)d_in[5];
    const float* bi     = (const float*)d_in[6];
    const float* Whrz   = (const float*)d_in[7];
    const float* Whn    = (const float*)d_in[8];
    const float* bhn    = (const float*)d_in[9];
    bf16* wsb = (bf16*)d_ws;

    prep_weights<<<(WS_ELEMS + 255) / 256, 256, 0, stream>>>(U1, U2, Whrz, Whn, wsb);

    // 16 waves/block * 16 rows/wave = 256 rows per block
    gru_fused<<<BATCH / (16 * WAVES), 32 * WAVES, 0, stream>>>(inputs, h1, h2, wsb,
                                                              Wi, bi, bhn, (float*)d_out);
}
